// MultiHeadAttention_41077067219537
// MI455X (gfx1250) — matrix operand order repaired and checked by reading, compiled, not run on hardware
//
#include <hip/hip_runtime.h>
#include <hip/hip_bf16.h>
#include <stdint.h>

typedef __attribute__((ext_vector_type(16))) __bf16 v16bf;
typedef __attribute__((ext_vector_type(8)))  float  v8f;

// ---------- gfx1250 async global->LDS path (guarded) ----------
#if defined(__gfx1250__) && __has_builtin(__builtin_amdgcn_global_load_async_to_lds_b128)
#define HAVE_ASYNC_LDS 1
#else
#define HAVE_ASYNC_LDS 0
#endif

#if HAVE_ASYNC_LDS
// builtin signature (from hipcc diagnostic): (int4 AS1*, int4 AS3*, imm i32, imm i32)
typedef __attribute__((__vector_size__(4 * sizeof(int)))) int i32x4;
typedef __attribute__((address_space(1))) i32x4* glb_b128_ptr;
typedef __attribute__((address_space(3))) i32x4* lds_b128_ptr;

__device__ __forceinline__ void async_b128(const void* gsrc, void* ldst) {
  // global_load_async_to_lds_b128: 16 bytes per lane, tracked by ASYNCcnt
  __builtin_amdgcn_global_load_async_to_lds_b128(
      (glb_b128_ptr)(uintptr_t)gsrc,
      (lds_b128_ptr)(unsigned)(uintptr_t)ldst,   // low 32 bits = LDS offset
      0, 0);
}
__device__ __forceinline__ void wait_async0() {
#if __has_builtin(__builtin_amdgcn_s_wait_asynccnt)
  __builtin_amdgcn_s_wait_asynccnt(0);
#else
  asm volatile("s_wait_asynccnt 0" ::: "memory");
#endif
}
#else
__device__ __forceinline__ void wait_async0() {}
#endif

// ---------- helpers ----------

__device__ __forceinline__ __bf16 f2bf(float f) {
  unsigned u = __builtin_bit_cast(unsigned, f);
  u = (u + 0x7FFFu + ((u >> 16) & 1u)) >> 16;          // round-to-nearest-even
  unsigned short s = (unsigned short)u;
  return __builtin_bit_cast(__bf16, s);
}

__device__ __forceinline__ v8f wmma_bf16(v16bf a, v16bf b, v8f c) {
  // v_wmma_f32_16x16x32_bf16: D = A(16x32) * B(32x16) + C(16x16 f32)
  return __builtin_amdgcn_wmma_f32_16x16x32_bf16(false, a, false, b, (short)0, c,
                                                 false, false);
}

// A-matrix fragment (16x32 bf16, row-major tile in LDS with row stride ld).
// ISA: lane m = lane&15; lanes 0-15 hold K=0..7,16..23; lanes 16-31 hold
// K=8..15,24..31; each VGPR packs 2 consecutive K.
__device__ __forceinline__ v16bf load_a_frag(const __bf16* base, int ld) {
  const int lane = threadIdx.x & 31;
  const int m  = lane & 15;
  const int kb = (lane >> 4) << 3;                     // 0 or 8
  v16bf a;
#pragma unroll
  for (int v = 0; v < 8; ++v) {
    const int k = ((v & 4) << 2) + kb + ((v & 3) << 1);
    a[2 * v]     = base[m * ld + k];
    a[2 * v + 1] = base[m * ld + k + 1];
  }
  return a;
}

// B-matrix fragment (32x16 bf16). Element (k,n) at base[k*sk + n*sn].
// ISA: lane n = lane&15; lanes 0-15 hold K=0..7,16..23; lanes 16-31 K=8..15,24..31.
__device__ __forceinline__ v16bf load_b_frag(const __bf16* base, int sk, int sn) {
  const int lane = threadIdx.x & 31;
  const int n  = lane & 15;
  const int kb = (lane >> 4) << 3;                     // 0 or 8
  v16bf b;
#pragma unroll
  for (int v = 0; v < 8; ++v) {
    const int k = ((v & 4) << 2) + kb + ((v & 3) << 1);
    b[2 * v]     = base[k * sk + n * sn];
    b[2 * v + 1] = base[(k + 1) * sk + n * sn];
  }
  return b;
}

// ---------- GEMM: Out[M,N] = A[M,K] * W[K,N] + bias ----------
// BM=128, BN=128, BK=32. 8 waves in a 4(m) x 2(n) grid: each wave computes
// 32 rows x 64 cols = 2 A-frags x 4 B-frags -> 8 WMMAs per K-step.

template <bool A_F32, bool OUT_F32>
__global__ __launch_bounds__(256) void gemm_bias_kernel(
    const void* __restrict__ Aptr, const float* __restrict__ W,
    const float* __restrict__ bias, void* __restrict__ Out,
    int M, int N, int K) {
  constexpr int BM = 128, BN = 128, BK = 32;
  __shared__ __bf16 As[BM][BK + 8];      // 128 x 40 (80 B rows, 16B-aligned chunks)
  __shared__ __bf16 Bs[BK][BN + 8];      // 32 x 136

  const int tid  = threadIdx.x;
  const int wave = tid >> 5;
  const int lane = tid & 31;
  const int wm = wave & 3;               // 4 waves over M (32 rows each)
  const int wn = wave >> 2;              // 2 waves over N (64 cols each)
  const int m0 = blockIdx.y * BM;
  const int n0 = blockIdx.x * BN;

  v8f acc[2][4] = {};

  for (int k0 = 0; k0 < K; k0 += BK) {
    __syncthreads();
    // stage A tile (128x32)
    if (A_F32) {
      const float* Af = (const float*)Aptr;
#pragma unroll
      for (int t = 0; t < 4; ++t) {                     // 1024 float4 chunks
        const int i = tid + t * 256;
        const int r = i >> 3, c = (i & 7) << 2;
        const float4 f = *(const float4*)&Af[(size_t)(m0 + r) * K + k0 + c];
        As[r][c + 0] = f2bf(f.x); As[r][c + 1] = f2bf(f.y);
        As[r][c + 2] = f2bf(f.z); As[r][c + 3] = f2bf(f.w);
      }
    } else {
      const __bf16* Ab = (const __bf16*)Aptr;
#pragma unroll
      for (int t = 0; t < 2; ++t) {                     // 512 uint4 chunks
        const int i = tid + t * 256;
        const int r = i >> 2, c = (i & 3) << 3;
        *(uint4*)&As[r][c] = *(const uint4*)&Ab[(size_t)(m0 + r) * K + k0 + c];
      }
    }
    // stage W tile (32x128, always f32 in global)
#pragma unroll
    for (int t = 0; t < 4; ++t) {                       // 1024 float4 chunks
      const int i = tid + t * 256;
      const int r = i >> 5, c = (i & 31) << 2;
      const float4 f = *(const float4*)&W[(size_t)(k0 + r) * N + n0 + c];
      Bs[r][c + 0] = f2bf(f.x); Bs[r][c + 1] = f2bf(f.y);
      Bs[r][c + 2] = f2bf(f.z); Bs[r][c + 3] = f2bf(f.w);
    }
    __syncthreads();

    const v16bf a0 = load_a_frag(&As[wm * 32][0], BK + 8);
    const v16bf a1 = load_a_frag(&As[wm * 32 + 16][0], BK + 8);
#pragma unroll
    for (int nt = 0; nt < 4; ++nt) {
      const v16bf b = load_b_frag(&Bs[0][wn * 64 + nt * 16], BN + 8, 1);
      acc[0][nt] = wmma_bf16(a0, b, acc[0][nt]);
      acc[1][nt] = wmma_bf16(a1, b, acc[1][nt]);
    }
  }

  // epilogue: C/D layout — lane holds column n, 8 row-registers
  const int n      = lane & 15;
  const int rowoff = (lane >> 4) << 3;
#pragma unroll
  for (int mt = 0; mt < 2; ++mt)
#pragma unroll
    for (int nt = 0; nt < 4; ++nt) {
      const int gn = n0 + wn * 64 + nt * 16 + n;
      const float bv = bias[gn];
#pragma unroll
      for (int r = 0; r < 8; ++r) {
        const int gm = m0 + wm * 32 + mt * 16 + rowoff + r;
        const float v = acc[mt][nt][r] + bv;
        if (OUT_F32)
          ((float*)Out)[(size_t)gm * N + gn] = v;
        else
          ((__bf16*)Out)[(size_t)gm * N + gn] = f2bf(v);
      }
    }
}

// ---------- Flash attention (causal, online softmax) ----------
// Grid: (S/128, H, B). 8 waves; wave w owns queries [q0+16w, q0+16w+16) x HD=64.
// K/V blocks of 64 keys double-buffered in LDS with async global->LDS copies.

__global__ __launch_bounds__(256) void flash_attn_kernel(
    const __bf16* __restrict__ Q, const __bf16* __restrict__ Kt,
    const __bf16* __restrict__ V, __bf16* __restrict__ O, int S, int D) {
  constexpr int HD = 64, LDK = HD + 8;   // 144 B rows, 16B-aligned chunks
  __shared__ __bf16 Qs[128][LDK];
  __shared__ __bf16 Ks[2][64][LDK];
  __shared__ __bf16 Vs[2][64][LDK];
  __shared__ __bf16 Ps[8][16][LDK];      // per-wave P staging (layout transpose)

  const int tid = threadIdx.x, wave = tid >> 5, lane = tid & 31;
  const int q0 = blockIdx.x * 128;
  const int h = blockIdx.y, b = blockIdx.z;
  const size_t rowbase = (size_t)b * S;
  const int coff = h * HD;

  // ---- stage Q tile (128 x 64 bf16) ----
#if HAVE_ASYNC_LDS
#pragma unroll
  for (int t = 0; t < 4; ++t) {                      // 1024 chunks of 8 bf16
    const int i = tid + t * 256;
    const int r = i >> 3, c = (i & 7) << 3;
    async_b128(&Q[(rowbase + q0 + r) * D + coff + c], &Qs[r][c]);
  }
#else
  for (int i = tid; i < 128 * HD; i += 256) {
    const int r = i >> 6, c = i & 63;
    Qs[r][c] = Q[(rowbase + q0 + r) * D + coff + c];
  }
#endif

  auto stage_kv = [&](int buf, int kb) {
#if HAVE_ASYNC_LDS
#pragma unroll
    for (int t = 0; t < 2; ++t) {                    // 512 chunks of 8 bf16
      const int i = tid + t * 256;
      const int r = i >> 3, c = (i & 7) << 3;
      const size_t g = (rowbase + kb + r) * (size_t)D + coff + c;
      async_b128(&Kt[g], &Ks[buf][r][c]);
      async_b128(&V[g],  &Vs[buf][r][c]);
    }
#else
    for (int i = tid; i < 64 * HD; i += 256) {
      const int r = i >> 6, c = i & 63;
      const size_t g = (rowbase + kb + r) * (size_t)D + coff + c;
      Ks[buf][r][c] = Kt[g];
      Vs[buf][r][c] = V[g];
    }
#endif
  };

  stage_kv(0, 0);
  wait_async0();
  __syncthreads();

  const v16bf aq0 = load_a_frag(&Qs[wave * 16][0], LDK);
  const v16bf aq1 = load_a_frag(&Qs[wave * 16][32], LDK);

  v8f o[4] = {};
  float mrun[8], lrun[8];
#pragma unroll
  for (int r = 0; r < 8; ++r) { mrun[r] = -1e30f; lrun[r] = 0.0f; }

  const int qrow_lane = q0 + wave * 16 + ((lane >> 4) << 3);
  const int kend = q0 + 128;                // causal: only keys < q0+128 matter

  int buf = 0;
  for (int kb = 0; kb < kend; kb += 64) {
    // prefetch next K/V block into the other buffer while we compute
    if (kb + 64 < kend) stage_kv(buf ^ 1, kb + 64);

    // scores: S = Q (16xHD) * K^T (HDx64), HD contracted in two K=32 steps
    v8f sc[4];
#pragma unroll
    for (int nt = 0; nt < 4; ++nt) {
      const v16bf bk0 = load_b_frag(&Ks[buf][nt * 16][0], 1, LDK);   // transposed
      const v16bf bk1 = load_b_frag(&Ks[buf][nt * 16][32], 1, LDK);
      v8f t = {};
      t = wmma_bf16(aq0, bk0, t);
      t = wmma_bf16(aq1, bk1, t);
      sc[nt] = t;
    }

    // scale + causal mask + row max
    float rmax[8];
#pragma unroll
    for (int r = 0; r < 8; ++r) rmax[r] = -1e30f;
#pragma unroll
    for (int nt = 0; nt < 4; ++nt) {
      const int kg = kb + nt * 16 + (lane & 15);
#pragma unroll
      for (int r = 0; r < 8; ++r) {
        float v = sc[nt][r] * 0.125f;        // 1/sqrt(64)
        if (kg > qrow_lane + r) v = -1e30f;
        sc[nt][r] = v;
        rmax[r] = fmaxf(rmax[r], v);
      }
    }
#pragma unroll
    for (int mask = 1; mask < 16; mask <<= 1)
#pragma unroll
      for (int r = 0; r < 8; ++r)
        rmax[r] = fmaxf(rmax[r], __shfl_xor(rmax[r], mask, 32));

    // online softmax update
    float alpha[8];
#pragma unroll
    for (int r = 0; r < 8; ++r) {
      const float mn = fmaxf(mrun[r], rmax[r]);
      alpha[r] = __expf(mrun[r] - mn);
      mrun[r] = mn;
    }
#pragma unroll
    for (int nt = 0; nt < 4; ++nt)
#pragma unroll
      for (int r = 0; r < 8; ++r) o[nt][r] *= alpha[r];

    float rsum[8] = {0, 0, 0, 0, 0, 0, 0, 0};
#pragma unroll
    for (int nt = 0; nt < 4; ++nt)
#pragma unroll
      for (int r = 0; r < 8; ++r) {
        const float p = __expf(sc[nt][r] - mrun[r]);
        sc[nt][r] = p;
        rsum[r] += p;
      }
#pragma unroll
    for (int mask = 1; mask < 16; mask <<= 1)
#pragma unroll
      for (int r = 0; r < 8; ++r) rsum[r] += __shfl_xor(rsum[r], mask, 32);
#pragma unroll
    for (int r = 0; r < 8; ++r) lrun[r] = lrun[r] * alpha[r] + rsum[r];

    // C-layout -> A-fragment layout via per-wave LDS (DS ops in-order per wave)
#pragma unroll
    for (int nt = 0; nt < 4; ++nt)
#pragma unroll
      for (int r = 0; r < 8; ++r)
        Ps[wave][((lane >> 4) << 3) + r][nt * 16 + (lane & 15)] = f2bf(sc[nt][r]);

    const v16bf ap0 = load_a_frag(&Ps[wave][0][0], LDK);
    const v16bf ap1 = load_a_frag(&Ps[wave][0][32], LDK);

    // O += P (16x64) * V (64xHD), keys contracted in two K=32 steps
#pragma unroll
    for (int nt = 0; nt < 4; ++nt) {
      const v16bf bv0 = load_b_frag(&Vs[buf][0][nt * 16], LDK, 1);
      const v16bf bv1 = load_b_frag(&Vs[buf][32][nt * 16], LDK, 1);
      o[nt] = wmma_bf16(ap0, bv0, o[nt]);
      o[nt] = wmma_bf16(ap1, bv1, o[nt]);
    }

    wait_async0();       // our async fills of buf^1 have landed
    __syncthreads();     // everyone done reading buf, fills visible to all
    buf ^= 1;
  }

  // normalize and store attn output as bf16 in [B*S, D] (head h = cols h*64..)
#pragma unroll
  for (int r = 0; r < 8; ++r) {
    const float inv = 1.0f / lrun[r];
    const size_t grow = (rowbase + qrow_lane + r) * (size_t)D + coff;
#pragma unroll
    for (int nt = 0; nt < 4; ++nt)
      O[grow + nt * 16 + (lane & 15)] = f2bf(o[nt][r] * inv);
  }
}

// ---------- host ----------

extern "C" void kernel_launch(void* const* d_in, const int* in_sizes, int n_in,
                              void* d_out, int out_size, void* d_ws, size_t ws_size,
                              hipStream_t stream) {
  (void)in_sizes; (void)n_in; (void)out_size; (void)ws_size;
  const int B = 2, S = 2048, D = 1024, H = 16;
  const int M = B * S;

  const float* x  = (const float*)d_in[0];
  const float* Wq = (const float*)d_in[1];
  const float* bq = (const float*)d_in[2];
  const float* Wk = (const float*)d_in[3];
  const float* bk = (const float*)d_in[4];
  const float* Wv = (const float*)d_in[5];
  const float* bv = (const float*)d_in[6];
  const float* Wo = (const float*)d_in[7];
  const float* bo = (const float*)d_in[8];

  __bf16* Qb = (__bf16*)d_ws;                 // 4096x1024 bf16 each (8 MB)
  __bf16* Kb = Qb + (size_t)M * D;
  __bf16* Vb = Kb + (size_t)M * D;
  __bf16* Ab = Vb + (size_t)M * D;            // attention output (bf16)

  dim3 gg(D / 128, M / 128);                   // (8, 32)
  gemm_bias_kernel<true,  false><<<gg, 256, 0, stream>>>(x,  Wq, bq, Qb, M, D, D);
  gemm_bias_kernel<true,  false><<<gg, 256, 0, stream>>>(x,  Wk, bk, Kb, M, D, D);
  gemm_bias_kernel<true,  false><<<gg, 256, 0, stream>>>(x,  Wv, bv, Vb, M, D, D);

  dim3 ga(S / 128, H, B);                      // (16, 16, 2)
  flash_attn_kernel<<<ga, 256, 0, stream>>>(Qb, Kb, Vb, Ab, S, D);

  gemm_bias_kernel<false, true><<<gg, 256, 0, stream>>>(Ab, Wo, bo, d_out, M, D, D);
}
